// SLM_31688268710463
// MI455X (gfx1250) — matrix operand order repaired and checked by reading, not compiled, not run on hardware
//
#include <hip/hip_runtime.h>
#include <math.h>
#include <stdint.h>

// ---------------- model constants (fixed by reference) ----------------
#define VV   32000
#define DD   1024
#define SS   1024
#define HH   16
#define NLAY 8
#define DKK  64
#define BB   2
#define DFF  4096
#define M2   (BB * SS)          // 2048 total tokens

typedef __attribute__((ext_vector_type(16))) _Float16 v16h;
typedef __attribute__((ext_vector_type(8)))  float    v8f;

union FragH { v16h v; _Float16 h[16]; };

// ---------------------------------------------------------------------
// Embedding: x[b,s,:] = emb_tab[index[b,s]] + pos_tab[s]
// ---------------------------------------------------------------------
__global__ __launch_bounds__(256) void embed_kernel(
    const int* __restrict__ idx, const float* __restrict__ emb,
    const float* __restrict__ pos, float* __restrict__ x)
{
    int t  = blockIdx.x * 256 + threadIdx.x;      // [0, B*S*D)
    int d  = t & (DD - 1);
    int bs = t >> 10;                             // token id [0, 2048)
    int s  = bs & (SS - 1);
    int tok = idx[bs];
    x[t] = emb[(size_t)tok * DD + d] + pos[(size_t)s * DD + d];
}

// ---------------------------------------------------------------------
// LayerNorm: one block per row of 1024
// ---------------------------------------------------------------------
__global__ __launch_bounds__(256) void ln_kernel(
    const float* __restrict__ x, const float* __restrict__ g,
    const float* __restrict__ b, float* __restrict__ out)
{
    __shared__ float red[256];
    int row = blockIdx.x, t = threadIdx.x;
    const float* xr = x + (size_t)row * DD;
    float v[4], s = 0.f;
#pragma unroll
    for (int i = 0; i < 4; ++i) { v[i] = xr[t + 256 * i]; s += v[i]; }
    red[t] = s; __syncthreads();
    for (int o = 128; o > 0; o >>= 1) { if (t < o) red[t] += red[t + o]; __syncthreads(); }
    float mu = red[0] * (1.f / DD);
    __syncthreads();
    float s2 = 0.f;
#pragma unroll
    for (int i = 0; i < 4; ++i) { float d = v[i] - mu; s2 += d * d; }
    red[t] = s2; __syncthreads();
    for (int o = 128; o > 0; o >>= 1) { if (t < o) red[t] += red[t + o]; __syncthreads(); }
    float rs = rsqrtf(red[0] * (1.f / DD) + 1e-5f);
#pragma unroll
    for (int i = 0; i < 4; ++i) {
        int c = t + 256 * i;
        out[(size_t)row * DD + c] = (v[i] - mu) * rs * g[c] + b[c];
    }
}

// ---------------------------------------------------------------------
// Core WMMA GEMM tile body: C[128x64] = A[M,K] @ W[K,N] (row-major, fp32
// in memory, f16 staged in LDS, f32 WMMA accumulation). 128 threads =
// 4 waves; wave w computes rows [32w,32w+32) x 64 cols:
// 2 A-frags x 4 B-frags = 8 wmma per K-step (B-frag reused twice).
// flags: 1=+bias[N]  2=+resid[M,N]  4=relu  8=store f16
// ---------------------------------------------------------------------
#define BM  128
#define LDP 40   // 32 + 8 pad (f16 elements)

__device__ __forceinline__ void gemm_tile_body(
    const float* __restrict__ A, const float* __restrict__ W,
    const float* __restrict__ bias, const float* __restrict__ resid,
    void* __restrict__ Cv, int N, int K, int bx, int by, int flags)
{
    __shared__ _Float16 As[BM * LDP];
    __shared__ _Float16 Bs[64 * LDP];
    int tid = threadIdx.x, wave = tid >> 5, lane = tid & 31;
    int hi = lane >> 4, l15 = lane & 15;
    int row0 = by * BM, col0 = bx * 64;

    v8f acc[2][4] = {};
    for (int k0 = 0; k0 < K; k0 += 32) {
        // A tile 128x32 -> f16 LDS
        for (int i = tid; i < BM * 32; i += 128) {
            int r = i >> 5, c = i & 31;
            As[r * LDP + c] = (_Float16)A[(size_t)(row0 + r) * K + k0 + c];
        }
        // W tile 32x64 -> f16 LDS, transposed (n-major, k contiguous)
        for (int i = tid; i < 32 * 64; i += 128) {
            int kk = i >> 6, n = i & 63;
            Bs[n * LDP + kk] = (_Float16)W[(size_t)(k0 + kk) * N + col0 + n];
        }
        __syncthreads();

        FragH a[2];
#pragma unroll
        for (int mt = 0; mt < 2; ++mt) {
            int am = wave * 32 + mt * 16 + l15;     // A-frag: lane = row
#pragma unroll
            for (int e = 0; e < 16; ++e) {
                int kk = e + 8 * hi + ((e >= 8) ? 8 : 0);   // ISA 16-bit A layout
                a[mt].h[e] = As[am * LDP + kk];
            }
        }
#pragma unroll
        for (int nt = 0; nt < 4; ++nt) {
            FragH bf;                                // B-frag: lane = col, k = e + 8*hi + (e>=8 ? 8 : 0)
#pragma unroll
            for (int e = 0; e < 16; ++e)
                bf.h[e] = Bs[(nt * 16 + l15) * LDP + e + 8 * hi + ((e >= 8) ? 8 : 0)];
#pragma unroll
            for (int mt = 0; mt < 2; ++mt)
                acc[mt][nt] = __builtin_amdgcn_wmma_f32_16x16x32_f16(
                    false, a[mt].v, false, bf.v, (short)0, acc[mt][nt], false, false);
        }
        __syncthreads();
    }
    // epilogue (C/D layout: row = 8*hi + r, col = lane&15)
#pragma unroll
    for (int mt = 0; mt < 2; ++mt) {
#pragma unroll
        for (int nt = 0; nt < 4; ++nt) {
#pragma unroll
            for (int r = 0; r < 8; ++r) {
                int gr = row0 + wave * 32 + mt * 16 + 8 * hi + r;
                int gc = col0 + nt * 16 + l15;
                float o = acc[mt][nt][r];
                if (flags & 1) o += bias[gc];
                if (flags & 2) o += resid[(size_t)gr * N + gc];
                if (flags & 4) o = fmaxf(o, 0.f);
                size_t ix = (size_t)gr * N + gc;
                if (flags & 8) ((_Float16*)Cv)[ix] = (_Float16)o;
                else           ((float*)Cv)[ix]    = o;
            }
        }
    }
}

__global__ __launch_bounds__(128) void gemm_rm_kernel(
    const float* A, const float* W, const float* bias, const float* resid,
    float* C, int N, int K, int flags)
{
    gemm_tile_body(A, W, bias, resid, (void*)C, N, K, blockIdx.x, blockIdx.y, flags);
}

// per-(b,h) projection GEMM, f16 output: C[z] = A[b] @ W[h],  z = b*H + h
__global__ __launch_bounds__(128) void gemm_heads_kernel(
    const float* A, const float* W, _Float16* C)
{
    int z = blockIdx.z, b = z >> 4, h = z & 15;
    gemm_tile_body(A + (size_t)b * SS * DD,
                   W + (size_t)h * DD * DKK,
                   nullptr, nullptr,
                   (void*)(C + (size_t)z * SS * DKK),
                   DKK, DD, blockIdx.x, blockIdx.y, 8);
}

// ---------------------------------------------------------------------
// Flash-style causal attention. grid = (S/64, B*H), block = 128 (4 waves).
// q,k,v: [B,H,S,DK] f16 (stored by projection GEMMs). out: [B,S,H*DK] f32.
// Q/K/V tiles are contiguous f16 blocks -> copied into LDS with
// global_load_async_to_lds_b128 (ASYNCcnt) instead of VGPR round-trips.
// scale = D^-0.5 (reference scales by d_model, not dk) = 1/32.
// ---------------------------------------------------------------------
#define ATP 72   // 64 + 8 pad (f16 elements)

__device__ __forceinline__ void async_copy_row16(uint32_t lds_off, const _Float16* gsrc)
{
    uint64_t ga = (uint64_t)(uintptr_t)gsrc;
    asm volatile("global_load_async_to_lds_b128 %0, %1, off"
                 :: "v"(lds_off), "v"(ga) : "memory");
}

__device__ __forceinline__ float rmax16(float x) {
#pragma unroll
    for (int m = 1; m < 16; m <<= 1) x = fmaxf(x, __shfl_xor(x, m, 32));
    return x;
}
__device__ __forceinline__ float rsum16(float x) {
#pragma unroll
    for (int m = 1; m < 16; m <<= 1) x += __shfl_xor(x, m, 32);
    return x;
}

__global__ __launch_bounds__(128) void attn_kernel(
    const _Float16* __restrict__ q, const _Float16* __restrict__ k,
    const _Float16* __restrict__ v, float* __restrict__ attn)
{
    const float SCALE = 0.03125f;   // 1/sqrt(1024)
    int z = blockIdx.y, b = z >> 4, h = z & 15;
    int q0 = blockIdx.x * 64;
    int tid = threadIdx.x, wave = tid >> 5, lane = tid & 31;
    int hi = lane >> 4, l15 = lane & 15;

    const _Float16* qb = q + (size_t)z * SS * DKK;
    const _Float16* kb = k + (size_t)z * SS * DKK;
    const _Float16* vb = v + (size_t)z * SS * DKK;

    __shared__ _Float16 Qs[64 * ATP];
    __shared__ _Float16 Ks[32 * ATP];
    __shared__ _Float16 Vs[32 * ATP];
    __shared__ _Float16 Ps[4][16 * LDP];   // per-wave P scratch

    uint32_t qso = (uint32_t)(uintptr_t)&Qs[0];
    uint32_t kso = (uint32_t)(uintptr_t)&Ks[0];
    uint32_t vso = (uint32_t)(uintptr_t)&Vs[0];

    // async copy Q tile (64 rows x 128B, contiguous) into padded LDS
    {
        const _Float16* qt = qb + (size_t)q0 * DKK;
#pragma unroll
        for (int c = tid; c < 512; c += 128) {          // 512 chunks of 16B
            int r = c >> 3, sc0 = (c & 7) * 8;
            async_copy_row16(qso + (uint32_t)(r * ATP + sc0) * 2,
                             qt + r * DKK + sc0);
        }
    }
    asm volatile("s_wait_asynccnt 0" ::: "memory");
    __syncthreads();

    // q fragments for this wave's 16 rows (dk chunks 0-31 and 32-63)
    FragH aq0, aq1;
    int qm = wave * 16 + l15;
#pragma unroll
    for (int e = 0; e < 16; ++e) {
        int kk = e + 8 * hi + ((e >= 8) ? 8 : 0);
        aq0.h[e] = Qs[qm * ATP + kk];
        aq1.h[e] = Qs[qm * ATP + 32 + kk];
    }

    v8f accO[4] = {};
    float mst[8], lst[8];
#pragma unroll
    for (int r = 0; r < 8; ++r) { mst[r] = -1e30f; lst[r] = 0.f; }

    int ktmax = (q0 >> 5) + 2;              // causal: keys <= q0+63
    for (int kt = 0; kt < ktmax; ++kt) {
        // async copy K,V tiles (32 rows x 128B each, contiguous)
        const _Float16* ktile = kb + (size_t)kt * 32 * DKK;
        const _Float16* vtile = vb + (size_t)kt * 32 * DKK;
#pragma unroll
        for (int c = tid; c < 256; c += 128) {          // 256 chunks of 16B
            int r = c >> 3, sc0 = (c & 7) * 8;
            async_copy_row16(kso + (uint32_t)(r * ATP + sc0) * 2,
                             ktile + r * DKK + sc0);
            async_copy_row16(vso + (uint32_t)(r * ATP + sc0) * 2,
                             vtile + r * DKK + sc0);
        }
        asm volatile("s_wait_asynccnt 0" ::: "memory");
        __syncthreads();

        // scores: 2 sub-tiles of 16 keys, contraction over DK=64 (2 wmma each)
        v8f sc[2];
#pragma unroll
        for (int sub = 0; sub < 2; ++sub) {
            FragH bk0, bk1;
            int keyl = sub * 16 + l15;      // B-frag: lane = key col
#pragma unroll
            for (int e = 0; e < 16; ++e) {
                int dk0 = e + 8 * hi + ((e >= 8) ? 8 : 0);
                bk0.h[e] = Ks[keyl * ATP + dk0];
                bk1.h[e] = Ks[keyl * ATP + 32 + dk0];
            }
            v8f s = {};
            s = __builtin_amdgcn_wmma_f32_16x16x32_f16(false, aq0.v, false, bk0.v, (short)0, s, false, false);
            s = __builtin_amdgcn_wmma_f32_16x16x32_f16(false, aq1.v, false, bk1.v, (short)0, s, false, false);
#pragma unroll
            for (int r = 0; r < 8; ++r) {
                int qrow = q0 + wave * 16 + 8 * hi + r;
                int kcol = kt * 32 + sub * 16 + l15;
                float val = s[r] * SCALE;
                if (kcol > qrow) val = -1e30f;
                s[r] = val;
            }
            sc[sub] = s;
        }

        // online softmax update (row groups = 16 lanes per half)
#pragma unroll
        for (int r = 0; r < 8; ++r) {
            float mx   = rmax16(fmaxf(sc[0][r], sc[1][r]));
            float mnew = fmaxf(mst[r], mx);
            float p0 = __expf(sc[0][r] - mnew);
            float p1 = __expf(sc[1][r] - mnew);
            float corr = __expf(mst[r] - mnew);
            float rs = rsum16(p0 + p1);
            lst[r] = lst[r] * corr + rs;
            mst[r] = mnew;
#pragma unroll
            for (int nt = 0; nt < 4; ++nt) accO[nt][r] *= corr;
            sc[0][r] = p0; sc[1][r] = p1;
        }

        // P (C-layout) -> wave-private LDS -> A-fragment
#pragma unroll
        for (int r = 0; r < 8; ++r) {
            int m = 8 * hi + r;
            Ps[wave][m * LDP + l15]      = (_Float16)sc[0][r];
            Ps[wave][m * LDP + 16 + l15] = (_Float16)sc[1][r];
        }
        asm volatile("s_wait_dscnt 0" ::: "memory");
        FragH ap;
#pragma unroll
        for (int e = 0; e < 16; ++e) {
            int kk = e + 8 * hi + ((e >= 8) ? 8 : 0);
            ap.h[e] = Ps[wave][l15 * LDP + kk];
        }
        // accO += P @ V  (contraction over 32 keys)
#pragma unroll
        for (int nt = 0; nt < 4; ++nt) {
            FragH bv;
#pragma unroll
            for (int e = 0; e < 16; ++e)
                bv.h[e] = Vs[(e + 8 * hi + ((e >= 8) ? 8 : 0)) * ATP + nt * 16 + l15];
            accO[nt] = __builtin_amdgcn_wmma_f32_16x16x32_f16(
                false, ap.v, false, bv.v, (short)0, accO[nt], false, false);
        }
        __syncthreads();
    }

    // write concat-head output
#pragma unroll
    for (int nt = 0; nt < 4; ++nt) {
#pragma unroll
        for (int r = 0; r < 8; ++r) {
            int qrow = q0 + wave * 16 + 8 * hi + r;
            int dk = nt * 16 + l15;
            attn[((size_t)(b * SS + qrow)) * DD + h * DKK + dk] = accO[nt][r] / lst[r];
        }
    }
}

// ---------------------------------------------------------------------
// Orchestration
// ---------------------------------------------------------------------
extern "C" void kernel_launch(void* const* d_in, const int* in_sizes, int n_in,
                              void* d_out, int out_size, void* d_ws, size_t ws_size,
                              hipStream_t stream)
{
    (void)in_sizes; (void)n_in; (void)out_size; (void)ws_size;
    const int*   index  = (const int*)  d_in[0];
    const float* emb    = (const float*)d_in[1];
    const float* pos    = (const float*)d_in[2];
    const float* wq     = (const float*)d_in[3];
    const float* wk     = (const float*)d_in[4];
    const float* wv     = (const float*)d_in[5];
    const float* wo_w   = (const float*)d_in[6];
    const float* wo_b   = (const float*)d_in[7];
    const float* w1     = (const float*)d_in[8];
    const float* b1     = (const float*)d_in[9];
    const float* w2     = (const float*)d_in[10];
    const float* b2     = (const float*)d_in[11];
    const float* ln1_g  = (const float*)d_in[12];
    const float* ln1_b  = (const float*)d_in[13];
    const float* ln2_g  = (const float*)d_in[14];
    const float* ln2_b  = (const float*)d_in[15];
    const float* lnf_g  = (const float*)d_in[16];
    const float* lnf_b  = (const float*)d_in[17];
    const float* lm_w   = (const float*)d_in[18];
    const float* lm_b   = (const float*)d_in[19];
    float* out = (float*)d_out;

    // workspace layout
    float* ws   = (float*)d_ws;
    const size_t TOK = (size_t)M2 * DD;      // 2048*1024
    float* x    = ws;                        // residual stream (f32)
    float* hbuf = x    + TOK;                // LN output (f32)
    float* ab   = hbuf + TOK;                // attention out, concat heads (f32)
    float* mid  = ab   + TOK;                // FFN intermediate [2048,4096] (f32)
    _Float16* qh = (_Float16*)(mid + (size_t)M2 * DFF);   // [B,H,S,DK] f16
    _Float16* kh = qh + TOK;
    _Float16* vh = kh + TOK;

    embed_kernel<<<(M2 * DD) / 256, 256, 0, stream>>>(index, emb, pos, x);

    const size_t wqkvL = (size_t)HH * DD * DKK;   // per-layer qkv weight stride
    const size_t woL   = (size_t)DD * DD;
    const size_t w1L   = (size_t)DD * DFF;

    for (int L = 0; L < NLAY; ++L) {
        ln_kernel<<<M2, 256, 0, stream>>>(x, ln1_g + L * DD, ln1_b + L * DD, hbuf);

        dim3 gh(1, SS / BM, BB * HH);
        gemm_heads_kernel<<<gh, 128, 0, stream>>>(hbuf, wq + (size_t)L * wqkvL, qh);
        gemm_heads_kernel<<<gh, 128, 0, stream>>>(hbuf, wk + (size_t)L * wqkvL, kh);
        gemm_heads_kernel<<<gh, 128, 0, stream>>>(hbuf, wv + (size_t)L * wqkvL, vh);

        attn_kernel<<<dim3(SS / 64, BB * HH), 128, 0, stream>>>(qh, kh, vh, ab);

        // x = x + attn @ wo + wo_b
        gemm_rm_kernel<<<dim3(DD / 64, M2 / BM), 128, 0, stream>>>(
            ab, wo_w + (size_t)L * woL, wo_b + L * DD, x, x, DD, DD, 1 | 2);

        ln_kernel<<<M2, 256, 0, stream>>>(x, ln2_g + L * DD, ln2_b + L * DD, hbuf);

        // mid = relu(h @ w1 + b1)
        gemm_rm_kernel<<<dim3(DFF / 64, M2 / BM), 128, 0, stream>>>(
            hbuf, w1 + (size_t)L * w1L, b1 + (size_t)L * DFF, nullptr, mid, DFF, DD, 1 | 4);

        // x = x + mid @ w2 + b2
        gemm_rm_kernel<<<dim3(DD / 64, M2 / BM), 128, 0, stream>>>(
            mid, w2 + (size_t)L * w1L, b2 + L * DD, x, x, DD, DFF, 1 | 2);
    }

    ln_kernel<<<M2, 256, 0, stream>>>(x, lnf_g, lnf_b, hbuf);

    // logits = h @ lm_w + lm_b  (N = 32000 = 500 * 64)
    gemm_rm_kernel<<<dim3(VV / 64, M2 / BM), 128, 0, stream>>>(
        hbuf, lm_w, lm_b, nullptr, out, VV, DD, 1);
}
